// Transformer_67791763800229
// MI455X (gfx1250) — matrix operand order repaired and checked by reading, not compiled, not run on hardware
//
#include <hip/hip_runtime.h>
#include <hip/hip_bf16.h>
#include <math.h>

// ---------------------------------------------------------------------------
// CDNA5 (gfx1250) wave32 WMMA types & helpers
// ---------------------------------------------------------------------------
typedef __attribute__((ext_vector_type(8)))  _Float16 h8;
typedef __attribute__((ext_vector_type(16))) _Float16 v16h;
typedef __attribute__((ext_vector_type(8)))  float    v8f;

__device__ __forceinline__ v16h cat8(h8 lo, h8 hi) {
  return __builtin_shufflevector(lo, hi, 0,1,2,3,4,5,6,7,8,9,10,11,12,13,14,15);
}

__device__ __forceinline__ v8f wmma16(v16h a, v16h b, v8f c) {
  // D(16x16 f32) = A(16x32 f16) * B(32x16 f16) + C
  return __builtin_amdgcn_wmma_f32_16x16x32_f16(
      /*neg_a=*/false, a, /*neg_b=*/false, b,
      /*c_mod=*/(short)0, c, /*reuse_a=*/false, /*reuse_b=*/false);
}

// gfx1250 async global->LDS copy path (ASYNCcnt), with safe fallback.
#if defined(__has_builtin)
#if __has_builtin(__builtin_amdgcn_global_load_async_to_lds_b128)
#define HAVE_ASYNC_LDS 1
#endif
#endif

#if defined(HAVE_ASYNC_LDS)
typedef int i4 __attribute__((ext_vector_type(4)));
typedef __attribute__((address_space(1))) i4 gi4;   // global-side int4
typedef __attribute__((address_space(3))) i4 li4;   // LDS-side int4
#define ASYNC_WAIT0() asm volatile("s_wait_asynccnt 0x0" ::: "memory")
#endif

// ---------------------------------------------------------------------------
// GEMM: C[M,N] = A[M,K] @ W[N,K]^T + bias, optional exact GeLU.
// 256 threads = 8 waves; block tile 128x64x32; wave tile 32x32 (2x2 WMMA).
// ---------------------------------------------------------------------------
#define BM 128
#define BN 64
#define BK 32

__global__ __launch_bounds__(256) void k_gemm(
    const float* __restrict__ A, const float* __restrict__ W,
    const float* __restrict__ bias, float* __restrict__ C,
    int M, int N, int K, int act)
{
  __shared__ alignas(32) _Float16 As[BM * BK];
  __shared__ alignas(32) _Float16 Bs[BN * BK];

  const int tid  = threadIdx.x;
  const int wave = tid >> 5;
  const int lane = tid & 31;
  const int bm = blockIdx.y * BM;
  const int bn = blockIdx.x * BN;
  const int wm = (wave >> 1) * 32;   // 4 waves down M
  const int wn = (wave & 1) * 32;    // 2 waves across N
  const int l15 = lane & 15;
  const int lko = (lane >> 4) * 16;  // A-fragment K offset pattern
  const int kb  = (lane >> 4) * 16;  // B-fragment K base

  // tile staging assignments
  const int ar = tid >> 1;           // 0..127  A row
  const int ac = (tid & 1) * 16;     // A col half-slab
  const int br = tid >> 2;           // 0..63   B row (weight row n)
  const int bc = (tid & 3) * 8;      // B col slab

  v8f acc[2][2] = {};

  for (int k0 = 0; k0 < K; k0 += BK) {
    const float* ap = A + (size_t)(bm + ar) * K + k0 + ac;
    #pragma unroll
    for (int i = 0; i < 16; ++i) As[ar * BK + ac + i] = (_Float16)ap[i];

    const float* wp = W + (size_t)(bn + br) * K + k0 + bc;
    #pragma unroll
    for (int i = 0; i < 8; ++i) Bs[br * BK + bc + i] = (_Float16)wp[i];

    if (k0 + BK < K) {               // prefetch next K-slab -> global_prefetch_b8
      __builtin_prefetch(ap + BK, 0, 1);
      __builtin_prefetch(wp + BK, 0, 1);
    }
    __syncthreads();

    v16h af[2], bf[2];
    #pragma unroll
    for (int i = 0; i < 2; ++i) {
      const _Float16* p = &As[(wm + i * 16 + l15) * BK + lko];
      af[i] = cat8(*(const h8*)p, *(const h8*)(p + 8));
    }
    #pragma unroll
    for (int j = 0; j < 2; ++j) {
      const _Float16* p = &Bs[(wn + j * 16 + l15) * BK + kb];
      bf[j] = cat8(*(const h8*)p, *(const h8*)(p + 8));
    }
    #pragma unroll
    for (int i = 0; i < 2; ++i)
      #pragma unroll
      for (int j = 0; j < 2; ++j)
        acc[i][j] = wmma16(af[i], bf[j], acc[i][j]);
    __syncthreads();
  }

  #pragma unroll
  for (int i = 0; i < 2; ++i) {
    #pragma unroll
    for (int j = 0; j < 2; ++j) {
      const int col = bn + wn + j * 16 + l15;
      const float bv = bias ? bias[col] : 0.f;
      #pragma unroll
      for (int r = 0; r < 8; ++r) {
        const int row = bm + wm + i * 16 + r + ((lane >> 4) << 3);
        float v = acc[i][j][r] + bv;
        if (act) v = 0.5f * v * (1.f + erff(v * 0.70710678118654752f));
        C[(size_t)row * N + col] = v;
      }
    }
  }
}

// ---------------------------------------------------------------------------
// Fused attention: one workgroup (4 waves) per (b, h, 16-row q tile).
// scores(16xSkv) via WMMA -> mask -> softmax in LDS -> ctx = P @ V via WMMA.
// V slabs are staged into LDS with coalesced b128 (async-to-LDS on gfx1250
// toolchains that expose the builtin; plain vector copy otherwise).
// Layout: q/k/v/out are [B, Seq, 1024] with head h in cols h*64 .. h*64+63.
// mode 0: key mask (toks[b,key]!=0). mode 1: causal && toks[b,q]!=0.
// ---------------------------------------------------------------------------
__global__ __launch_bounds__(128) void k_attn(
    const float* __restrict__ Q, const float* __restrict__ Km,
    const float* __restrict__ Vm, const int* __restrict__ toks,
    float* __restrict__ O, int Sq, int Skv, int mode)
{
  const int qt = blockIdx.x, h = blockIdx.y, b = blockIdx.z;
  __shared__ alignas(32) float    sc[16 * 512];
  __shared__ alignas(32) _Float16 qs[16 * 64];
  __shared__ alignas(16) float    Vs[32 * 64];   // one 32-key V slab (8 KB)
  __shared__ float red[128], rmax[16], rsum[16];

  const int tid = threadIdx.x, wave = tid >> 5, lane = tid & 31;
  const int l15 = lane & 15;
  const int lko = (lane >> 4) * 16;
  const int kb  = (lane >> 4) * 16;

  // stage q tile (pre-scaled by 1/sqrt(64))
  {
    const int r = tid >> 3, c0 = (tid & 7) * 8;
    const float* qp = Q + (size_t)(b * Sq + qt * 16 + r) * 1024 + h * 64 + c0;
    #pragma unroll
    for (int i = 0; i < 8; ++i) qs[r * 64 + c0 + i] = (_Float16)(qp[i] * 0.125f);
  }
  __syncthreads();

  v16h af[2];
  #pragma unroll
  for (int kk = 0; kk < 2; ++kk) {
    const _Float16* p = &qs[l15 * 64 + kk * 32 + lko];
    af[kk] = cat8(*(const h8*)p, *(const h8*)(p + 8));
  }

  // ---- scores ----
  const int nkt = Skv >> 4;
  for (int kt = wave; kt < nkt; kt += 4) {
    v8f acc = {};
    const int key = kt * 16 + l15;
    #pragma unroll
    for (int kk = 0; kk < 2; ++kk) {
      const float* kp = Km + (size_t)(b * Skv + key) * 1024 + h * 64 + kk * 32 + kb;
      v16h bf;
      #pragma unroll
      for (int i = 0; i < 16; ++i) bf[i] = (_Float16)kp[i];
      acc = wmma16(af[kk], bf, acc);
    }
    #pragma unroll
    for (int r = 0; r < 8; ++r) {
      const int m = r + ((lane >> 4) << 3);
      const int qrow = qt * 16 + m;
      const bool ok = (mode == 0)
          ? (toks[b * Skv + key] != 0)
          : ((toks[b * Sq + qrow] != 0) && (qrow >= key));
      sc[m * 512 + key] = ok ? acc[r] : -1.0e9f;
    }
  }
  __syncthreads();

  // ---- softmax (8 threads per row) ----
  {
    const int row = tid >> 3, sub = tid & 7;
    float pm = -3.4e38f;
    for (int j = sub; j < Skv; j += 8) pm = fmaxf(pm, sc[row * 512 + j]);
    red[tid] = pm; __syncthreads();
    if (sub == 0) {
      float m = red[tid];
      #pragma unroll
      for (int t = 1; t < 8; ++t) m = fmaxf(m, red[tid + t]);
      rmax[row] = m;
    }
    __syncthreads();
    const float mx = rmax[row];
    float ps = 0.f;
    for (int j = sub; j < Skv; j += 8) {
      const float e = __expf(sc[row * 512 + j] - mx);
      sc[row * 512 + j] = e;
      ps += e;
    }
    red[tid] = ps; __syncthreads();
    if (sub == 0) {
      float s = 0.f;
      #pragma unroll
      for (int t = 0; t < 8; ++t) s += red[tid + t];
      rsum[row] = s;
    }
    __syncthreads();
    const float inv = 1.f / rsum[row];
    for (int j = sub; j < Skv; j += 8) sc[row * 512 + j] *= inv;
  }
  __syncthreads();

  // ---- ctx = P @ V : wave `wave` owns 16 columns of dh ----
  {
    const int vc = wave * 16 + l15;      // dh column within head (0..63)
    v8f acc = {};
    for (int kc = 0; kc < Skv; kc += 32) {
      // coalesced stage of V[kc..kc+31][h*64 .. h*64+63] into LDS
      {
        const int r = tid >> 2;          // 0..31 key row in slab
        const int c = (tid & 3) * 16;    // 0/16/32/48
        const float* gp = Vm + (size_t)(b * Skv + kc + r) * 1024 + h * 64 + c;
        float* lp = &Vs[r * 64 + c];
#if defined(HAVE_ASYNC_LDS)
        #pragma unroll
        for (int q4 = 0; q4 < 4; ++q4)
          __builtin_amdgcn_global_load_async_to_lds_b128(
              (gi4*)(gp + q4 * 4), (li4*)(lp + q4 * 4), 0, 0);
        ASYNC_WAIT0();
#else
        #pragma unroll
        for (int q4 = 0; q4 < 4; ++q4)
          *(float4*)(lp + q4 * 4) = *(const float4*)(gp + q4 * 4);
#endif
      }
      __syncthreads();

      v16h pa, vb;
      const float* sp = &sc[l15 * 512 + kc + lko];
      #pragma unroll
      for (int i = 0; i < 8; ++i) { pa[i] = (_Float16)sp[i]; pa[i + 8] = (_Float16)sp[8 + i]; }
      #pragma unroll
      for (int i = 0; i < 16; ++i) vb[i] = (_Float16)Vs[(kb + i) * 64 + vc];
      acc = wmma16(pa, vb, acc);
      __syncthreads();                    // Vs reused next iteration
    }
    #pragma unroll
    for (int r = 0; r < 8; ++r) {
      const int m = r + ((lane >> 4) << 3);
      O[(size_t)(b * Sq + qt * 16 + m) * 1024 + h * 64 + vc] = acc[r];
    }
  }
}

// ---------------------------------------------------------------------------
// Elementwise kernels
// ---------------------------------------------------------------------------
__global__ __launch_bounds__(256) void k_embed(
    const int* __restrict__ toks, const float* __restrict__ emb,
    float* __restrict__ out)
{
  const int row = blockIdx.x;
  const float* e = emb + (size_t)toks[row] * 1024;
  float* o = out + (size_t)row * 1024;
  for (int i = threadIdx.x; i < 1024; i += 256) o[i] = e[i];
}

__global__ __launch_bounds__(256) void k_rope(float* __restrict__ x, int Ssz)
{
  const int row = blockIdx.x;
  const int s = row % Ssz;
  float* p = x + (size_t)row * 1024;
  for (int i = threadIdx.x; i < 512; i += 256) {
    const float theta = powf(10000.f, -(float)i / 512.f);
    const float ang = (float)s * theta;
    const float sn = sinf(ang), cs = cosf(ang);
    const float x1 = p[2 * i], x2 = p[2 * i + 1];
    p[2 * i]     = x1 * cs - x2 * sn;
    p[2 * i + 1] = x1 * sn + x2 * cs;
  }
}

// x[row,:] += layernorm(a[row,:]) * g + b   (population variance, eps=1e-5)
__global__ __launch_bounds__(256) void k_lnres(
    const float* __restrict__ a, const float* __restrict__ g,
    const float* __restrict__ bta, float* __restrict__ x)
{
  const int row = blockIdx.x;
  const int tid = threadIdx.x;
  const float* ap = a + (size_t)row * 1024;
  float* xp = x + (size_t)row * 1024;
  __shared__ float s1[256], s2[256];
  float lsum = 0.f, lsq = 0.f;
  for (int i = tid; i < 1024; i += 256) { const float v = ap[i]; lsum += v; lsq += v * v; }
  s1[tid] = lsum; s2[tid] = lsq; __syncthreads();
  for (int off = 128; off > 0; off >>= 1) {
    if (tid < off) { s1[tid] += s1[tid + off]; s2[tid] += s2[tid + off]; }
    __syncthreads();
  }
  const float mean = s1[0] * (1.f / 1024.f);
  const float var  = s2[0] * (1.f / 1024.f) - mean * mean;
  const float inv  = rsqrtf(var + 1e-5f);
  for (int i = tid; i < 1024; i += 256)
    xp[i] += (ap[i] - mean) * inv * g[i] + bta[i];
}

// ---------------------------------------------------------------------------
// Orchestration
// ---------------------------------------------------------------------------
extern "C" void kernel_launch(void* const* d_in, const int* in_sizes, int n_in,
                              void* d_out, int out_size, void* d_ws, size_t ws_size,
                              hipStream_t stream) {
  const int Bn = 2, Ss = 512, Ts = 512, Dn = 1024, Hn = 16, Ln = 4, Vn = 32000;

  const int*   src     = (const int*)d_in[0];
  const int*   tgt     = (const int*)d_in[1];
  const float* emb_enc = (const float*)d_in[2];
  const float* emb_dec = (const float*)d_in[3];
  const float* eWq = (const float*)d_in[4];
  const float* eWk = (const float*)d_in[5];
  const float* eWv = (const float*)d_in[6];
  const float* eWo = (const float*)d_in[7];
  const float* ebo = (const float*)d_in[8];
  const float* eW1 = (const float*)d_in[9];
  const float* eb1 = (const float*)d_in[10];
  const float* eW2 = (const float*)d_in[11];
  const float* eb2 = (const float*)d_in[12];
  const float* eg1 = (const float*)d_in[13];
  const float* eB1 = (const float*)d_in[14];
  const float* eg2 = (const float*)d_in[15];
  const float* eB2 = (const float*)d_in[16];
  const float* dWq = (const float*)d_in[17];
  const float* dWk = (const float*)d_in[18];
  const float* dWv = (const float*)d_in[19];
  const float* dWo = (const float*)d_in[20];
  const float* dbo = (const float*)d_in[21];
  const float* dW1 = (const float*)d_in[22];
  const float* db1 = (const float*)d_in[23];
  const float* dW2 = (const float*)d_in[24];
  const float* db2 = (const float*)d_in[25];
  const float* dg1 = (const float*)d_in[26];
  const float* dB1 = (const float*)d_in[27];
  const float* dg2 = (const float*)d_in[28];
  const float* dB2 = (const float*)d_in[29];
  const float* dg3 = (const float*)d_in[30];
  const float* dB3 = (const float*)d_in[31];
  const float* oW  = (const float*)d_in[32];
  const float* ob  = (const float*)d_in[33];

  const size_t MSD = (size_t)Bn * Ss * Dn;  // 1,048,576 floats
  float* ws   = (float*)d_ws;
  float* x    = ws;            // encoder activations [B,S,D]
  float* y    = x + MSD;       // decoder activations [B,T,D]
  float* qb   = y + MSD;
  float* kbuf = qb + MSD;
  float* vbuf = kbuf + MSD;
  float* ctx  = vbuf + MSD;
  float* prj  = ctx + MSD;
  float* ffb  = prj + MSD;     // [B,S,4D] = 4*MSD

  const int Mrows = Bn * Ss;   // == Bn*Ts == 1024

  auto GEMM = [&](const float* A, const float* W, const float* bias, float* C,
                  int M, int N, int K, int act) {
    dim3 g(N / BN, M / BM);
    k_gemm<<<g, 256, 0, stream>>>(A, W, bias, C, M, N, K, act);
  };

  k_embed<<<Mrows, 256, 0, stream>>>(src, emb_enc, x);
  k_embed<<<Bn * Ts, 256, 0, stream>>>(tgt, emb_dec, y);

  // -------- encoder --------
  for (int i = 0; i < Ln; ++i) {
    const size_t wo = (size_t)i * Dn * Dn, vo = (size_t)i * Dn;
    const size_t w1o = (size_t)i * 4 * Dn * Dn, b1o = (size_t)i * 4 * Dn;
    GEMM(x, eWq + wo, nullptr, qb,   Mrows, Dn, Dn, 0);
    GEMM(x, eWk + wo, nullptr, kbuf, Mrows, Dn, Dn, 0);
    GEMM(x, eWv + wo, nullptr, vbuf, Mrows, Dn, Dn, 0);
    k_rope<<<Mrows, 256, 0, stream>>>(kbuf, Ss);
    k_rope<<<Mrows, 256, 0, stream>>>(vbuf, Ss);
    k_attn<<<dim3(Ss / 16, Hn, Bn), 128, 0, stream>>>(qb, kbuf, vbuf, src, ctx, Ss, Ss, 0);
    GEMM(ctx, eWo + wo, ebo + vo, prj, Mrows, Dn, Dn, 0);
    k_lnres<<<Mrows, 256, 0, stream>>>(prj, eg1 + vo, eB1 + vo, x);
    GEMM(x,   eW1 + w1o, eb1 + b1o, ffb, Mrows, 4 * Dn, Dn, 1);
    GEMM(ffb, eW2 + w1o, eb2 + vo,  prj, Mrows, Dn, 4 * Dn, 0);
    k_lnres<<<Mrows, 256, 0, stream>>>(prj, eg2 + vo, eB2 + vo, x);
  }

  // -------- decoder --------
  for (int i = 0; i < Ln; ++i) {
    const size_t wo = (size_t)i * Dn * Dn, vo = (size_t)i * Dn;
    const size_t w1o = (size_t)i * 4 * Dn * Dn, b1o = (size_t)i * 4 * Dn;
    // self-attention (causal, tgt mask)
    GEMM(y, dWq + wo, nullptr, qb,   Mrows, Dn, Dn, 0);
    GEMM(y, dWk + wo, nullptr, kbuf, Mrows, Dn, Dn, 0);
    GEMM(y, dWv + wo, nullptr, vbuf, Mrows, Dn, Dn, 0);
    k_rope<<<Mrows, 256, 0, stream>>>(kbuf, Ts);
    k_rope<<<Mrows, 256, 0, stream>>>(vbuf, Ts);
    k_attn<<<dim3(Ts / 16, Hn, Bn), 128, 0, stream>>>(qb, kbuf, vbuf, tgt, ctx, Ts, Ts, 1);
    GEMM(ctx, dWo + wo, dbo + vo, prj, Mrows, Dn, Dn, 0);
    k_lnres<<<Mrows, 256, 0, stream>>>(prj, dg1 + vo, dB1 + vo, y);
    // cross-attention (keys/values from encoder output x, src mask)
    GEMM(y, dWq + wo, nullptr, qb,   Mrows, Dn, Dn, 0);
    GEMM(x, dWk + wo, nullptr, kbuf, Mrows, Dn, Dn, 0);
    GEMM(x, dWv + wo, nullptr, vbuf, Mrows, Dn, Dn, 0);
    k_rope<<<Mrows, 256, 0, stream>>>(kbuf, Ss);
    k_rope<<<Mrows, 256, 0, stream>>>(vbuf, Ss);
    k_attn<<<dim3(Ts / 16, Hn, Bn), 128, 0, stream>>>(qb, kbuf, vbuf, src, ctx, Ts, Ss, 0);
    GEMM(ctx, dWo + wo, dbo + vo, prj, Mrows, Dn, Dn, 0);
    k_lnres<<<Mrows, 256, 0, stream>>>(prj, dg2 + vo, dB2 + vo, y);
    // feed-forward
    GEMM(y,   dW1 + w1o, db1 + b1o, ffb, Mrows, 4 * Dn, Dn, 1);
    GEMM(ffb, dW2 + w1o, db2 + vo,  prj, Mrows, Dn, 4 * Dn, 0);
    k_lnres<<<Mrows, 256, 0, stream>>>(prj, dg3 + vo, dB3 + vo, y);
  }

  // -------- output logits --------
  GEMM(y, oW, ob, (float*)d_out, Mrows, Vn, Dn, 0);

  (void)in_sizes; (void)n_in; (void)out_size; (void)ws_size;
}
